// GPNNCell2_20641612824994
// MI455X (gfx1250) — matrix operand order repaired and checked by reading, not compiled, not run on hardware
//
#include <hip/hip_runtime.h>
#include <cstdint>
#include <cstddef>

typedef __bf16 bf16;
typedef __attribute__((ext_vector_type(16))) __bf16 v16bf;
typedef __attribute__((ext_vector_type(8)))  __bf16 v8bf;
typedef __attribute__((ext_vector_type(8)))  float  v8f;
typedef __attribute__((ext_vector_type(4)))  unsigned int v4u;
typedef __attribute__((ext_vector_type(8)))  int v8i;
typedef __attribute__((ext_vector_type(4)))  int v4i;

// ---------------- problem constants ----------------
// B=4, T=128, N=10, D=768
#define NN    10
#define DIM   768
#define M1    5120      /* B*T*N   node rows  */
#define M2    51200     /* B*T*N*N edge rows  */
#define TND   983040    /* T*N*D : batch stride for GraphNorm */

#if __has_builtin(__builtin_amdgcn_tensor_load_to_lds) && __has_builtin(__builtin_amdgcn_s_wait_tensorcnt)
#define USE_TDM 1
#else
#define USE_TDM 0
#endif

static __device__ __forceinline__ float gelu_exact(float x) {
    return 0.5f * x * (1.0f + erff(x * 0.70710678118654752440f));
}

static __device__ __forceinline__ float blockReduceSum256(float v) {
    __shared__ float sb[256];
    int tid = threadIdx.x;
    sb[tid] = v;
    __syncthreads();
#pragma unroll
    for (int s = 128; s > 0; s >>= 1) {
        if (tid < s) sb[tid] += sb[tid + s];
        __syncthreads();
    }
    float r = sb[0];
    __syncthreads();
    return r;
}

// ---------------- cast fp32 -> bf16 ----------------
__global__ void cast_f32_bf16_kernel(const float* __restrict__ s, bf16* __restrict__ d, size_t n) {
    size_t i = (size_t)blockIdx.x * blockDim.x + threadIdx.x;
    if (i < n) d[i] = (bf16)s[i];
}

// ------------- cast + transpose 768x768 weight: d[n*768+k] = (bf16)s[k*768+n] -------------
__global__ __launch_bounds__(256) void cast_transpose_bf16_kernel(
    const float* __restrict__ s, bf16* __restrict__ d) {
    __shared__ float t[32][33];
    int bx = blockIdx.x * 32, by = blockIdx.y * 32;   // bx: n-range, by: k-range
    int tx = threadIdx.x & 31, ty = threadIdx.x >> 5; // 32 x 8
#pragma unroll
    for (int r = 0; r < 32; r += 8) t[ty + r][tx] = s[(size_t)(by + ty + r) * DIM + bx + tx];
    __syncthreads();
#pragma unroll
    for (int r = 0; r < 32; r += 8) d[(size_t)(bx + ty + r) * DIM + by + tx] = (bf16)t[tx][ty + r];
}

// ---------------- TDM tile load: <rows> x 32 bf16, row stride 768 elems ----------------
// LDS padding: pad_interval=16 DWORDs (64B = one 32-elem row), pad_amount=4 DWORDs (16B)
// -> LDS row pitch 40 bf16 elems (16B aligned chunks, bank-spread).
#define LP 40
#define TILEA (128 * LP) /* elems per LDS A-tile buffer */
#define TILEB (256 * LP) /* elems per LDS B-tile buffer */

#if USE_TDM
static __device__ __forceinline__ void tdm_load_tile(unsigned int lds_addr,
                                                     unsigned long long gaddr, int rows) {
    v4u g0;
    g0[0] = 1u;                               // count=1, user descriptor
    g0[1] = lds_addr;                         // LDS byte address
    g0[2] = (unsigned int)gaddr;              // global addr [31:0]
    g0[3] = (unsigned int)((gaddr >> 32) & 0x01FFFFFFu) | (2u << 30); // [56:32] | type=2
    v8i g1;
    g1[0] = 0x06D10000;  // wg_mask=0 | data_size=1(2B)<<16 | pad_en<<20 | pad_int=3<<22 | pad_amt=3<<25
    g1[1] = (int)(768u << 16);   // tensor_dim0[15:0] in [31:16]
    g1[2] = (int)(768u << 16);   // tensor_dim1[15:0] in [31:16]
    g1[3] = (int)(32u << 16);    // tile_dim0 = 32 in [31:16]
    g1[4] = rows;                // tile_dim1
    g1[5] = 768;                 // tensor_dim0_stride[31:0]
    g1[6] = (int)(768u << 16);   // tensor_dim1_stride[15:0] in [31:16]
    g1[7] = 0;
    v4i z4; z4[0] = 0; z4[1] = 0; z4[2] = 0; z4[3] = 0;
#if __clang_major__ >= 23
    v8i z8; z8[0] = 0; z8[1] = 0; z8[2] = 0; z8[3] = 0; z8[4] = 0; z8[5] = 0; z8[6] = 0; z8[7] = 0;
    __builtin_amdgcn_tensor_load_to_lds(g0, g1, z4, z4, z8, 0);
#else
    __builtin_amdgcn_tensor_load_to_lds(g0, g1, z4, z4, 0);
#endif
}
#endif

// ---------------- bf16 WMMA GEMM, C = A(MxK) * BT(NxK)^T, K=N=768 ----------------
// BT is the pre-transposed weight (row n holds column n of W, K-contiguous).
// 128x256 block tile, K-step 32, 256 threads = 8 wave32 (2 in M x 4 in N),
// each wave: 4x4 grid of 16x16 v_wmma_f32_16x16x32_bf16 accumulators (64x64 patch).
union Frag16 { v16bf v; v8bf h[2]; };

template <int EPI>
__global__ __launch_bounds__(256) void gemm_bf16_kernel(
    const bf16* __restrict__ A, const bf16* __restrict__ BT,
    float* __restrict__ outF, bf16* __restrict__ outBF,
    const float* __restrict__ bias,
    const float* __restrict__ rowI, const float* __restrict__ rowJ) {
#if USE_TDM
    __shared__ bf16 As[2 * TILEA];
    __shared__ bf16 Bs[2 * TILEB];
#else
    __shared__ bf16 As[TILEA];
    __shared__ bf16 Bs[TILEB];
#endif
    const int tid  = threadIdx.x;
    const int lane = tid & 31;
    const int wid  = tid >> 5;
    const int wm   = wid & 1;   // wave position in M (2)
    const int wn   = wid >> 1;  // wave position in N (4)
    const int lm   = lane & 15;
    const int hf   = lane >> 4; // half-wave selects K-subrange
    const int row0 = blockIdx.x * 128;
    const int col0 = blockIdx.y * 256;

    v8f acc[4][4];
#pragma unroll
    for (int mi = 0; mi < 4; mi++)
#pragma unroll
        for (int ni = 0; ni < 4; ni++)
#pragma unroll
            for (int r = 0; r < 8; r++) acc[mi][ni][r] = 0.f;

#if USE_TDM
    int buf = 0;
    if (wid == 0) {
        tdm_load_tile((unsigned int)(uintptr_t)&As[0],
                      (unsigned long long)(uintptr_t)(A + (size_t)row0 * DIM), 128);
        tdm_load_tile((unsigned int)(uintptr_t)&Bs[0],
                      (unsigned long long)(uintptr_t)(BT + (size_t)col0 * DIM), 256);
    }
#endif

    for (int k0 = 0; k0 < DIM; k0 += 32) {
#if USE_TDM
        if (wid == 0) __builtin_amdgcn_s_wait_tensorcnt((short)0);
        __syncthreads();
        if (wid == 0 && k0 + 32 < DIM) {
            int nb = buf ^ 1;
            tdm_load_tile((unsigned int)(uintptr_t)&As[nb * TILEA],
                          (unsigned long long)(uintptr_t)(A + (size_t)row0 * DIM + (k0 + 32)), 128);
            tdm_load_tile((unsigned int)(uintptr_t)&Bs[nb * TILEB],
                          (unsigned long long)(uintptr_t)(BT + (size_t)col0 * DIM + (k0 + 32)), 256);
        }
        const bf16* Ac = &As[buf * TILEA];
        const bf16* Bc = &Bs[buf * TILEB];
#else
        __syncthreads();
        if (k0 + 32 < DIM) {  // global_prefetch_b8 for next K tile
            __builtin_prefetch(&A[(size_t)(row0 + (tid >> 1)) * DIM + (k0 + 32)], 0, 0);
            __builtin_prefetch(&BT[(size_t)(col0 + tid) * DIM + (k0 + 32)], 0, 0);
        }
        // both tiles row-major (K contiguous), 16B chunks: A 2/thread, B 4/thread
#pragma unroll
        for (int q = 0; q < 2; q++) {
            int ch = tid * 2 + q;
            int r = ch >> 2, kc = ch & 3;
            *(v8bf*)(&As[r * LP + kc * 8]) = *(const v8bf*)(&A[(size_t)(row0 + r) * DIM + k0 + kc * 8]);
        }
#pragma unroll
        for (int q = 0; q < 4; q++) {
            int ch = tid * 4 + q;
            int r = ch >> 2, kc = ch & 3;
            *(v8bf*)(&Bs[r * LP + kc * 8]) = *(const v8bf*)(&BT[(size_t)(col0 + r) * DIM + k0 + kc * 8]);
        }
        __syncthreads();
        const bf16* Ac = &As[0];
        const bf16* Bc = &Bs[0];
#endif
        // A frag (16x32 bf16): lane = M row; half-wave hf: K base 0/8, +16 in v4..7
        Frag16 af[4], bfg[4];
#pragma unroll
        for (int mi = 0; mi < 4; mi++) {
            int r = wm * 64 + mi * 16 + lm;
            af[mi].h[0] = *(const v8bf*)(&Ac[r * LP + hf * 8]);
            af[mi].h[1] = *(const v8bf*)(&Ac[r * LP + hf * 8 + 16]);
        }
        // B frag (32x16 bf16): lane = N col; half-wave hf: K base 0/8, +16 in v4..7
#pragma unroll
        for (int ni = 0; ni < 4; ni++) {
            int c = wn * 64 + ni * 16 + lm;
            bfg[ni].h[0] = *(const v8bf*)(&Bc[c * LP + hf * 8]);
            bfg[ni].h[1] = *(const v8bf*)(&Bc[c * LP + hf * 8 + 16]);
        }
#pragma unroll
        for (int mi = 0; mi < 4; mi++)
#pragma unroll
            for (int ni = 0; ni < 4; ni++)
                acc[mi][ni] = __builtin_amdgcn_wmma_f32_16x16x32_bf16(
                    false, af[mi].v, false, bfg[ni].v, (short)0, acc[mi][ni], false, false);
#if USE_TDM
        buf ^= 1;
#endif
    }

    // epilogue: C/D layout: VGPR r -> row (hf*8 + r), lane lm -> col
#pragma unroll
    for (int mi = 0; mi < 4; mi++) {
#pragma unroll
        for (int ni = 0; ni < 4; ni++) {
#pragma unroll
            for (int r = 0; r < 8; r++) {
                int gr = row0 + wm * 64 + mi * 16 + hf * 8 + r;
                int gc = col0 + wn * 64 + ni * 16 + lm;
                float v = acc[mi][ni][r];
                size_t o = (size_t)gr * DIM + gc;
                if constexpr (EPI == 0) {
                    outF[o] = v;
                } else if constexpr (EPI == 1) {
                    int iRow = gr / NN;
                    int jRow = (gr / (NN * NN)) * NN + gr % NN;
                    v += bias[gc] + rowI[(size_t)iRow * DIM + gc] + rowJ[(size_t)jRow * DIM + gc];
                    outBF[o] = (bf16)gelu_exact(v);
                } else if constexpr (EPI == 2) {
                    outF[o] = v + bias[gc];
                } else if constexpr (EPI == 3) {
                    int jRow = (gr / (NN * NN)) * NN + gr % NN;
                    v += bias[gc] + rowJ[(size_t)jRow * DIM + gc];
                    outBF[o] = (bf16)v;
                }
            }
        }
    }
}

// ---------------- LayerNorm + GELU over rows of 768 (edge_features) ----------------
__global__ __launch_bounds__(256) void ln_gelu_kernel(
    const float* __restrict__ pre, const float* __restrict__ w, const float* __restrict__ b,
    float* __restrict__ outF, bf16* __restrict__ outBF) {
    int row = blockIdx.x, tid = threadIdx.x;
    const float* p = pre + (size_t)row * DIM;
    float v0 = p[tid], v1 = p[tid + 256], v2 = p[tid + 512];
    float mu = blockReduceSum256(v0 + v1 + v2) * (1.f / DIM);
    float d0 = v0 - mu, d1 = v1 - mu, d2 = v2 - mu;
    float var = blockReduceSum256(d0 * d0 + d1 * d1 + d2 * d2) * (1.f / DIM);
    float inv = rsqrtf(var + 1e-12f);
    size_t base = (size_t)row * DIM;
    float g0 = gelu_exact(d0 * inv * w[tid] + b[tid]);
    float g1 = gelu_exact(d1 * inv * w[tid + 256] + b[tid + 256]);
    float g2 = gelu_exact(d2 * inv * w[tid + 512] + b[tid + 512]);
    outF[base + tid] = g0;         outBF[base + tid] = (bf16)g0;
    outF[base + tid + 256] = g1;   outBF[base + tid + 256] = (bf16)g1;
    outF[base + tid + 512] = g2;   outBF[base + tid + 512] = (bf16)g2;
}

// ---------------- gate (sigmoid(ef@Wlink+blink)) * gelu(LN2(mv_pre)) summed over j ---
__global__ __launch_bounds__(256) void gate_agg_kernel(
    const float* __restrict__ mvpre, const float* __restrict__ ef,
    const float* __restrict__ ln2w, const float* __restrict__ ln2b,
    const float* __restrict__ wlink, const float* __restrict__ blink,
    float* __restrict__ agg) {
    int row2 = blockIdx.x, tid = threadIdx.x;
    int c0 = tid, c1 = tid + 256, c2 = tid + 512;
    float wl0 = wlink[c0], wl1 = wlink[c1], wl2 = wlink[c2];
    float w0 = ln2w[c0], w1 = ln2w[c1], w2 = ln2w[c2];
    float b0 = ln2b[c0], b1 = ln2b[c1], b2 = ln2b[c2];
    float bl = blink[0];
    float a0 = 0.f, a1 = 0.f, a2 = 0.f;
    for (int j = 0; j < NN; j++) {
        size_t er = (size_t)(row2 * NN + j) * DIM;
        float v0 = mvpre[er + c0], v1 = mvpre[er + c1], v2 = mvpre[er + c2];
        float e0 = ef[er + c0], e1 = ef[er + c1], e2 = ef[er + c2];
        float mu = blockReduceSum256(v0 + v1 + v2) * (1.f / DIM);
        float dot = blockReduceSum256(e0 * wl0 + e1 * wl1 + e2 * wl2);
        float d0 = v0 - mu, d1 = v1 - mu, d2 = v2 - mu;
        float var = blockReduceSum256(d0 * d0 + d1 * d1 + d2 * d2) * (1.f / DIM);
        float inv = rsqrtf(var + 1e-12f);
        float g = 1.f / (1.f + expf(-(dot + bl)));
        a0 += g * gelu_exact(d0 * inv * w0 + b0);
        a1 += g * gelu_exact(d1 * inv * w1 + b1);
        a2 += g * gelu_exact(d2 * inv * w2 + b2);
    }
    size_t base = (size_t)row2 * DIM;
    agg[base + c0] = a0; agg[base + c1] = a1; agg[base + c2] = a2;
}

// ---------------- MessageNorm + residual ----------------
__global__ __launch_bounds__(256) void msgnorm_kernel(
    const float* __restrict__ agg, const float* __restrict__ x,
    const float* __restrict__ res_scale, float* __restrict__ yF, bf16* __restrict__ yBF) {
    int row2 = blockIdx.x, tid = threadIdx.x;
    size_t base = (size_t)row2 * DIM;
    float a0 = agg[base + tid], a1 = agg[base + tid + 256], a2 = agg[base + tid + 512];
    float x0 = x[base + tid], x1 = x[base + tid + 256], x2 = x[base + tid + 512];
    float na = sqrtf(blockReduceSum256(a0 * a0 + a1 * a1 + a2 * a2));
    float nx = sqrtf(blockReduceSum256(x0 * x0 + x1 * x1 + x2 * x2));
    float sc = nx * res_scale[0] / fmaxf(na, 1e-12f);
    float y0 = a0 * sc + x0, y1 = a1 * sc + x1, y2 = a2 * sc + x2;
    yF[base + tid] = y0;        yBF[base + tid] = (bf16)y0;
    yF[base + tid + 256] = y1;  yBF[base + tid + 256] = (bf16)y1;
    yF[base + tid + 512] = y2;  yBF[base + tid + 512] = (bf16)y2;
}

// ---------------- GraphNorm over batch axis (B=4) + GELU ----------------
__global__ __launch_bounds__(256) void graphnorm_kernel(
    const float* __restrict__ yW, const float* __restrict__ ms,
    const float* __restrict__ w, const float* __restrict__ b, float* __restrict__ out) {
    size_t idx = (size_t)blockIdx.x * 256 + threadIdx.x;
    if (idx >= TND) return;
    int d = (int)(idx % DIM);
    float v0 = yW[idx], v1 = yW[idx + TND], v2 = yW[idx + 2 * (size_t)TND], v3 = yW[idx + 3 * (size_t)TND];
    float mean = (v0 + v1 + v2 + v3) * 0.25f;
    float m = ms[d];
    float o0 = v0 - m * mean, o1 = v1 - m * mean, o2 = v2 - m * mean, o3 = v3 - m * mean;
    float var = (o0 * o0 + o1 * o1 + o2 * o2 + o3 * o3) * 0.25f;
    float inv = rsqrtf(var + 1e-5f);
    float gw = w[d], gb = b[d];
    out[idx]                   = gelu_exact(o0 * inv * gw + gb);
    out[idx + TND]             = gelu_exact(o1 * inv * gw + gb);
    out[idx + 2 * (size_t)TND] = gelu_exact(o2 * inv * gw + gb);
    out[idx + 3 * (size_t)TND] = gelu_exact(o3 * inv * gw + gb);
}

// ---------------- host launch ----------------
extern "C" void kernel_launch(void* const* d_in, const int* in_sizes, int n_in,
                              void* d_out, int out_size, void* d_ws, size_t ws_size,
                              hipStream_t stream) {
    (void)in_sizes; (void)n_in; (void)out_size; (void)ws_size;
    const float* node_features   = (const float*)d_in[0];
    const float* global_features = (const float*)d_in[1];
    const float* be1    = (const float*)d_in[5];
    const float* be2    = (const float*)d_in[7];
    const float* ln1_w  = (const float*)d_in[8];
    const float* ln1_b  = (const float*)d_in[9];
    const float* Wlink  = (const float*)d_in[10];
    const float* blink  = (const float*)d_in[11];
    const float* bm     = (const float*)d_in[14];
    const float* bmerge = (const float*)d_in[16];
    const float* ln2_w  = (const float*)d_in[17];
    const float* ln2_b  = (const float*)d_in[18];
    const float* res_sc = (const float*)d_in[19];
    const float* bn     = (const float*)d_in[21];
    const float* gn_w   = (const float*)d_in[22];
    const float* gn_b   = (const float*)d_in[23];
    const float* gn_ms  = (const float*)d_in[24];

    float* outY  = (float*)d_out;                     // (B,T,N,D)
    float* outEF = (float*)d_out + (size_t)M1 * DIM;  // (B,T,N,N,D)

    const size_t M1D = (size_t)M1 * DIM, M2D = (size_t)M2 * DIM, DD = (size_t)DIM * DIM;
    char* wsp = (char*)d_ws;
    size_t off = 0;
    auto take = [&](size_t bytes) -> void* {
        void* p = wsp + off;
        off = (off + bytes + 255) & ~(size_t)255;
        return p;
    };
    bf16*  xbf  = (bf16*)take(M1D * 2);
    bf16*  big  = (bf16*)take(M2D * 2);   // G, later m
    bf16*  hbf  = (bf16*)take(M2D * 2);   // h
    bf16*  efbf = (bf16*)take(M2D * 2);   // edge_features bf16
    float* pre  = (float*)take(M2D * 4);  // pre-LN buffer (reused)
    bf16*  WbfT = (bf16*)take(8 * DD * 2); // transposed weights (N-major, K-contig)
    float* ipart = (float*)take(M1D * 4);
    float* jpart = (float*)take(M1D * 4);
    float* xwmh  = (float*)take(M1D * 4);
    float* agg   = (float*)take(M1D * 4);
    float* yf    = (float*)take(M1D * 4);
    bf16*  ybf   = (bf16*)take(M1D * 2);
    float* yW    = (float*)take(M1D * 4);

    dim3 blk(256);
    auto cg = [](size_t n) { return dim3((unsigned)((n + 255) / 256)); };

    // casts: activations -> bf16; weights -> transposed bf16
    cast_f32_bf16_kernel<<<cg(M1D), blk, 0, stream>>>(node_features, xbf, M1D);
    cast_f32_bf16_kernel<<<cg(M2D), blk, 0, stream>>>(global_features, big, M2D);
    const int widx[8] = {2, 3, 4, 6, 12, 13, 15, 20}; // We_g We_i We_j We2 Wm_h Wm_e Wmerge Wn
    dim3 tg(DIM / 32, DIM / 32);
    for (int k = 0; k < 8; k++)
        cast_transpose_bf16_kernel<<<tg, blk, 0, stream>>>((const float*)d_in[widx[k]], WbfT + k * DD);

    dim3 g1(M1 / 128, DIM / 256);  // 40 x 3
    dim3 g2(M2 / 128, DIM / 256);  // 400 x 3

    // small node GEMMs
    gemm_bf16_kernel<0><<<g1, blk, 0, stream>>>(xbf, WbfT + 1 * DD, ipart, nullptr, nullptr, nullptr, nullptr);
    gemm_bf16_kernel<0><<<g1, blk, 0, stream>>>(xbf, WbfT + 2 * DD, jpart, nullptr, nullptr, nullptr, nullptr);
    gemm_bf16_kernel<0><<<g1, blk, 0, stream>>>(xbf, WbfT + 4 * DD, xwmh, nullptr, nullptr, nullptr, nullptr);

    // h = gelu(G@We_g + i_part[i] + j_part[j] + be1)
    gemm_bf16_kernel<1><<<g2, blk, 0, stream>>>(big, WbfT + 0 * DD, nullptr, hbf, be1, ipart, jpart);
    // ef_pre = h@We2 + be2 ; then LN1+GELU -> edge_features (output) + bf16 copy
    gemm_bf16_kernel<2><<<g2, blk, 0, stream>>>(hbf, WbfT + 3 * DD, pre, nullptr, be2, nullptr, nullptr);
    ln_gelu_kernel<<<dim3(M2), blk, 0, stream>>>(pre, ln1_w, ln1_b, outEF, efbf);
    // m = ef@Wm_e + xWm_h[j] + bm  (bf16, reuse 'big')
    gemm_bf16_kernel<3><<<g2, blk, 0, stream>>>(efbf, WbfT + 5 * DD, nullptr, big, bm, nullptr, xwmh);
    // mv_pre = m@Wmerge + bmerge
    gemm_bf16_kernel<2><<<g2, blk, 0, stream>>>(big, WbfT + 6 * DD, pre, nullptr, bmerge, nullptr, nullptr);
    // agg = sum_j sigmoid(ef@Wlink+blink) * gelu(LN2(mv_pre))
    gate_agg_kernel<<<dim3(M1), blk, 0, stream>>>(pre, outEF, ln2_w, ln2_b, Wlink, blink, agg);
    // MessageNorm + residual
    msgnorm_kernel<<<dim3(M1), blk, 0, stream>>>(agg, node_features, res_sc, yf, ybf);
    // yW = y@Wn + bn
    gemm_bf16_kernel<2><<<g1, blk, 0, stream>>>(ybf, WbfT + 7 * DD, yW, nullptr, bn, nullptr, nullptr);
    // GraphNorm over batch + GELU -> output
    graphnorm_kernel<<<cg(TND), blk, 0, stream>>>(yW, gn_ms, gn_w, gn_b, outY);
}
